// ACCriticNSMsg_79250736546101
// MI455X (gfx1250) — matrix operand order repaired and checked by reading, not compiled, not run on hardware
//
#include <hip/hip_runtime.h>

typedef __attribute__((ext_vector_type(16))) _Float16 v16h;
typedef __attribute__((ext_vector_type(8)))  _Float16 v8h;
typedef __attribute__((ext_vector_type(8)))  float    v8f;

#define BS      64
#define T       256
#define NA      8
#define ML      16
#define OBSD    128
#define HID     64
#define IN_DIM  240          // 128 + 7*16
#define KPAD    256          // padded K for 8 x k32 WMMA steps
#define KTILES  (KPAD/32)
#define M_TILE  64           // tokens per block (16 rows per wave, 4 waves)
#define ASTRIDE 264          // K stride (halves) with pad; 528B row, 16B aligned
#define BSTRIDE 264
#define THREADS 128
#define SMEM_BYTES ((M_TILE*ASTRIDE + HID*BSTRIDE)*2)

union V16 { v16h v; v8h h[2]; };

struct Frags {
    V16 a;       // A 16x32 f16 fragment
    V16 b[4];    // B 32x16 f16 fragments for the 4 N-tiles
};

// Load all fragments for k-tile kt. arow = lane's A row base, bbase = lane's
// B column base (ldsB + col*BSTRIDE).
__device__ __forceinline__
void load_frags(const _Float16* __restrict__ arow,
                const _Float16* __restrict__ bbase,
                int kt, int khalf, Frags& f)
{
    f.a.h[0] = *(const v8h*)(arow + kt*32 +      khalf*8);
    f.a.h[1] = *(const v8h*)(arow + kt*32 + 16 + khalf*8);
    #pragma unroll
    for (int nt = 0; nt < 4; ++nt) {
        const _Float16* bp = bbase + nt*16*BSTRIDE + kt*32 + khalf*8;
        f.b[nt].h[0] = *(const v8h*)(bp);
        f.b[nt].h[1] = *(const v8h*)(bp + 16);
    }
}

__global__ __launch_bounds__(THREADS)
void ACCriticNSMsg_kernel(const float* __restrict__ obs,
                          const float* __restrict__ messages,
                          const float* __restrict__ W1,
                          const float* __restrict__ b1,
                          const float* __restrict__ W2,
                          const float* __restrict__ b2,
                          float* __restrict__ out)
{
    extern __shared__ _Float16 smem[];
    _Float16* ldsA = smem;                   // [M_TILE][ASTRIDE] input tile
    _Float16* ldsB = smem + M_TILE*ASTRIDE;  // [HID][BSTRIDE]   W1[a]^T (N x K)

    const int a   = blockIdx.y;
    const int g0  = blockIdx.x * M_TILE;     // first flat token index (b*T + t)
    const int tid = threadIdx.x;

    // ---- stage A tile: gather obs ++ other-agents' messages(t-1) ++ zero pad,
    //      converting f32 -> f16 on the fly (coalesced: tid sweeps K) ----
    for (int e = tid; e < M_TILE*KPAD; e += THREADS) {
        const int row = e >> 8;
        const int k   = e & 255;
        const int g   = g0 + row;
        const int t   = g & (T - 1);
        float v = 0.f;
        if (k < OBSD) {
            v = obs[(size_t)(g*NA + a)*OBSD + k];
        } else if (k < IN_DIM && t > 0) {
            const int kk = k - OBSD;
            const int jj = kk >> 4;          // which "other agent" slot (0..6)
            const int m  = kk & 15;
            const int j  = jj + (jj >= a);   // skip self
            v = messages[(size_t)((g - 1)*NA + j)*ML + m];
        }
        ldsA[row*ASTRIDE + k] = (_Float16)v;
    }

    // ---- stage B tile: W1[a]^T with K padded to 256, f32 -> f16
    //      (tid sweeps H => coalesced 64-float rows of W1) ----
    for (int e = tid; e < HID*KPAD; e += THREADS) {
        const int h = e & (HID - 1);
        const int k = e >> 6;
        const float v = (k < IN_DIM)
            ? W1[(size_t)a*IN_DIM*HID + (size_t)k*HID + h] : 0.f;
        ldsB[h*BSTRIDE + k] = (_Float16)v;
    }
    __syncthreads();

    const int wave  = tid >> 5;
    const int lane  = tid & 31;
    const int col   = lane & 15;   // N column (and A row offset)
    const int khalf = lane >> 4;   // 0: K 0-7/16-23, 1: K 8-15/24-31
    const int mrow  = wave * 16;   // this wave's 16-token row block

    // accumulators (16x16 f32 C/D: lane=N, vgpr r = row M (+8 for hi lanes)),
    // pre-loaded with b1 broadcast (depends only on N)
    v8f acc[4];
    #pragma unroll
    for (int nt = 0; nt < 4; ++nt) {
        const float bv = b1[a*HID + nt*16 + col];
        #pragma unroll
        for (int r = 0; r < 8; ++r) acc[nt][r] = bv;
    }

    const _Float16* arow  = ldsA + (mrow + col)*ASTRIDE;
    const _Float16* bbase = ldsB + col*BSTRIDE;

    // ---- double-buffered fragment pipeline. The sched_barrier pins the
    //      prefetch DS loads for k-tile kt+1 ABOVE the WMMAs for kt, so both
    //      buffers are live at once (distinct VGPRs) and the pre-WMMA wait is
    //      a partial s_wait_dscnt instead of a full DS drain. ----
    Frags buf[2];
    load_frags(arow, bbase, 0, khalf, buf[0]);
    #pragma unroll
    for (int kt = 0; kt < KTILES; ++kt) {
        if (kt + 1 < KTILES)
            load_frags(arow, bbase, kt + 1, khalf, buf[(kt + 1) & 1]);
        __builtin_amdgcn_sched_barrier(0);   // keep prefetch issued before MMAs
        Frags& f = buf[kt & 1];
        #pragma unroll
        for (int nt = 0; nt < 4; ++nt) {
            acc[nt] = __builtin_amdgcn_wmma_f32_16x16x32_f16(
                false, f.a.v, false, f.b[nt].v, (short)0, acc[nt], false, false);
        }
    }

    // ---- fused stage 2: q = relu(h) . W2[a] + b2[a], reduced across the
    //      16-lane N groups with xor shuffles (wave32 safe) ----
    float w2v[4];
    #pragma unroll
    for (int nt = 0; nt < 4; ++nt) w2v[nt] = W2[a*HID + nt*16 + col];
    const float bias2 = b2[a];

    #pragma unroll
    for (int r = 0; r < 8; ++r) {
        float p = 0.f;
        #pragma unroll
        for (int nt = 0; nt < 4; ++nt)
            p += fmaxf(acc[nt][r], 0.f) * w2v[nt];
        p += __shfl_xor(p, 1, 32);
        p += __shfl_xor(p, 2, 32);
        p += __shfl_xor(p, 4, 32);
        p += __shfl_xor(p, 8, 32);
        if (col == 0) {
            const int g = g0 + mrow + khalf*8 + r;
            out[(size_t)g*NA + a] = p + bias2;
        }
    }
}

extern "C" void kernel_launch(void* const* d_in, const int* in_sizes, int n_in,
                              void* d_out, int out_size, void* d_ws, size_t ws_size,
                              hipStream_t stream) {
    const float* obs      = (const float*)d_in[0];
    const float* messages = (const float*)d_in[1];
    const float* W1       = (const float*)d_in[2];
    const float* b1       = (const float*)d_in[3];
    const float* W2       = (const float*)d_in[4];
    const float* b2       = (const float*)d_in[5];
    float* out = (float*)d_out;

    dim3 grid((BS*T)/M_TILE, NA);   // 256 token tiles x 8 agents = 2048 blocks
    dim3 block(THREADS);
    hipLaunchKernelGGL(ACCriticNSMsg_kernel, grid, block, SMEM_BYTES, stream,
                       obs, messages, W1, b1, W2, b2, out);
}
